// NanoBERT_2181843386999
// MI455X (gfx1250) — compile-verified, matrix operand order checked by reading, not run on hardware
//
#include <hip/hip_runtime.h>
#include <hip/hip_bf16.h>

// ---------------- types for WMMA / TDM ----------------
typedef __attribute__((ext_vector_type(16))) __bf16 v16bf;
typedef __attribute__((ext_vector_type(8)))  float  v8f;
typedef unsigned int v4u  __attribute__((ext_vector_type(4)));
typedef int          v8i_t __attribute__((ext_vector_type(8)));
typedef int          v4i_t __attribute__((ext_vector_type(4)));

union Frag { v16bf v; unsigned int u[8]; };

// pair base index inside a 16-bit A/B fragment (ISA 7.12.2 layout):
// VGPR i holds K = base, base+1 ; lanes 16-31 shifted by +8
__device__ __forceinline__ int pair_base(int i, int g) {
    return (i < 4 ? 2 * i : 16 + 2 * (i - 4)) + 8 * g;
}

__device__ __forceinline__ unsigned short f2bf(float f) {
    unsigned int u = __float_as_uint(f);
    unsigned int lsb = (u >> 16) & 1u;
    u += 0x7fffu + lsb;                 // round to nearest even
    return (unsigned short)(u >> 16);
}

__device__ __forceinline__ float gelu_exact(float x) {
    return 0.5f * x * (1.0f + erff(x * 0.70710678118654752f));
}

// low 32 bits of a generic pointer into LDS == LDS byte address (aperture rule)
__device__ __forceinline__ unsigned lds_addr_of(const void* p) {
    return (unsigned)(unsigned long long)(uintptr_t)p;
}

// ---- TDM: DMA a 64-row x 32-col bf16 tile (row stride = K elements) into LDS ----
// D# per CDNA5 ISA 8.3/8.4: 2-D tensor, data_size=2B, tile_dim0=32, tile_dim1=64.
__device__ __forceinline__ void tdm_load_tile_64x32(const unsigned short* gtile,
                                                    unsigned lds_off, int K, int Mtot) {
    unsigned long long ga = (unsigned long long)(uintptr_t)gtile;
    v4u g0;
    g0[0] = 1u;                                            // count=1, user mode
    g0[1] = lds_off;                                       // lds_addr
    g0[2] = (unsigned)(ga & 0xFFFFFFFFu);                  // global_addr[31:0]
    g0[3] = (unsigned)((ga >> 32) & 0x01FFFFFFu) | (2u << 30); // ga[56:32] | type=2
    v8i_t g1;
    g1[0] = 0x00010000;                                    // data_size=1 (2 bytes)
    g1[1] = (int)(((unsigned)K & 0xFFFFu) << 16);          // tensor_dim0[15:0]
    g1[2] = (int)((((unsigned)K >> 16) & 0xFFFFu) |
                  (((unsigned)Mtot & 0xFFFFu) << 16));     // td0[31:16] | td1[15:0]
    g1[3] = (int)((((unsigned)Mtot >> 16) & 0xFFFFu) |
                  (32u << 16));                            // td1[31:16] | tile_dim0=32
    g1[4] = 64;                                            // tile_dim1=64, tile_dim2=0
    g1[5] = K;                                             // tensor_dim0_stride[31:0]
    g1[6] = 0;
    g1[7] = 0;
    v4i_t z4 = {0, 0, 0, 0};
    v8i_t z8 = {0, 0, 0, 0, 0, 0, 0, 0};
    __builtin_amdgcn_tensor_load_to_lds(g0, g1, z4, z4, z8, 0);
}

// ---------------- weight repack (fp32 -> bf16, transposed to N x K) ----------------

// src (L,12,768,64) row-major  ->  dst (L, N=768, K=768), n = h*64+e
__global__ void repack_headed(const float* __restrict__ src, unsigned short* __restrict__ dst, int L) {
    size_t idx = (size_t)blockIdx.x * blockDim.x + threadIdx.x;
    size_t total = (size_t)L * 768 * 768;
    if (idx >= total) return;
    int k = (int)(idx % 768);
    size_t r = idx / 768;
    int n = (int)(r % 768);
    int l = (int)(r / 768);
    int hh = n >> 6, e = n & 63;
    dst[idx] = f2bf(src[(((size_t)l * 12 + hh) * 768 + k) * 64 + e]);
}

// src (L,K,N) row-major -> dst (L,N,K)
__global__ void repack_plain(const float* __restrict__ src, unsigned short* __restrict__ dst,
                             int L, int K, int N) {
    size_t idx = (size_t)blockIdx.x * blockDim.x + threadIdx.x;
    size_t total = (size_t)L * K * N;
    if (idx >= total) return;
    int k = (int)(idx % K);
    size_t r = idx / K;
    int n = (int)(r % N);
    int l = (int)(r / N);
    dst[idx] = f2bf(src[((size_t)l * K + k) * N + n]);
}

// ---------------- LayerNorm (768 wide), writes fp32 + bf16 ----------------

__global__ __launch_bounds__(256) void ln_kernel(const float* __restrict__ in,
                                                 float* __restrict__ outF,
                                                 unsigned short* __restrict__ outB,
                                                 const float* __restrict__ gamma,
                                                 const float* __restrict__ beta,
                                                 float eps) {
    __shared__ float red[256];
    int row = blockIdx.x;
    int t = threadIdx.x;
    const float* x = in + (size_t)row * 768;
    float v0 = x[t], v1 = x[t + 256], v2 = x[t + 512];
    red[t] = v0 + v1 + v2;
    __syncthreads();
    for (int st = 128; st > 0; st >>= 1) { if (t < st) red[t] += red[t + st]; __syncthreads(); }
    float mean = red[0] * (1.0f / 768.0f);
    __syncthreads();
    float d0 = v0 - mean, d1 = v1 - mean, d2 = v2 - mean;
    red[t] = d0 * d0 + d1 * d1 + d2 * d2;
    __syncthreads();
    for (int st = 128; st > 0; st >>= 1) { if (t < st) red[t] += red[t + st]; __syncthreads(); }
    float inv = rsqrtf(red[0] * (1.0f / 768.0f) + eps);
    float dv[3] = {d0, d1, d2};
    #pragma unroll
    for (int i = 0; i < 3; ++i) {
        int c = t + i * 256;
        float y = dv[i] * inv * gamma[c] + beta[c];
        outF[(size_t)row * 768 + c] = y;
        if (outB) outB[(size_t)row * 768 + c] = f2bf(y);
    }
}

// embedding gather + add pos + LayerNorm(eps=1e-12)
__global__ __launch_bounds__(256) void embed_ln_kernel(const int* __restrict__ xids,
                                                       const float* __restrict__ wemb,
                                                       const float* __restrict__ pemb,
                                                       const float* __restrict__ gamma,
                                                       const float* __restrict__ beta,
                                                       float* __restrict__ outF,
                                                       unsigned short* __restrict__ outB) {
    __shared__ float red[256];
    int row = blockIdx.x;          // b*S + s
    int s = row & 511;             // S = 512
    int t = threadIdx.x;
    int id = xids[row];
    const float* we = wemb + (size_t)id * 768;
    const float* pe = pemb + (size_t)s * 768;
    float v0 = we[t] + pe[t];
    float v1 = we[t + 256] + pe[t + 256];
    float v2 = we[t + 512] + pe[t + 512];
    red[t] = v0 + v1 + v2;
    __syncthreads();
    for (int st = 128; st > 0; st >>= 1) { if (t < st) red[t] += red[t + st]; __syncthreads(); }
    float mean = red[0] * (1.0f / 768.0f);
    __syncthreads();
    float d0 = v0 - mean, d1 = v1 - mean, d2 = v2 - mean;
    red[t] = d0 * d0 + d1 * d1 + d2 * d2;
    __syncthreads();
    for (int st = 128; st > 0; st >>= 1) { if (t < st) red[t] += red[t + st]; __syncthreads(); }
    float inv = rsqrtf(red[0] * (1.0f / 768.0f) + 1e-12f);
    float dv[3] = {d0, d1, d2};
    #pragma unroll
    for (int i = 0; i < 3; ++i) {
        int c = t + i * 256;
        float y = dv[i] * inv * gamma[c] + beta[c];
        outF[(size_t)row * 768 + c] = y;
        outB[(size_t)row * 768 + c] = f2bf(y);
    }
}

// ---------------- bf16 WMMA GEMM with TDM-staged, double-buffered A tiles ----------------
// Y[M,N] = act(X[M,K]*W[K,N] + bias (+res)).  X: M x K bf16.  Wt: N x K bf16.
// 128 threads = 4 waves; 64x64 tile; wave w owns N-slice w*16.
// Wave 0 drives the Tensor Data Mover: tile k+1 DMA overlaps tile-k WMMA.

__global__ __launch_bounds__(128) void gemm_bf16(const unsigned short* __restrict__ X,
                                                 const unsigned short* __restrict__ Wt,
                                                 const float* __restrict__ bias,
                                                 const float* __restrict__ resid,
                                                 float* __restrict__ outF,
                                                 unsigned short* __restrict__ outB,
                                                 int M, int N, int K, int act) {
    __shared__ unsigned short As[2][64 * 32];
    int tid = threadIdx.x;
    int wave = tid >> 5, lane = tid & 31;
    int g = lane >> 4, nl = lane & 15;
    int m0 = blockIdx.x * 64;
    int n0 = blockIdx.y * 64 + wave * 16;
    v8f acc[4] = {};
    const int nsteps = K >> 5;
    const unsigned short* xbase = X + (size_t)m0 * K;

    if (tid < 32)   // prime the pipeline: DMA tile 0
        tdm_load_tile_64x32(xbase, lds_addr_of(&As[0][0]), K, M);

    for (int kk = 0; kk < nsteps; ++kk) {
        if (tid < 32) {
            if (kk + 1 < nsteps) {
                tdm_load_tile_64x32(xbase + (size_t)(kk + 1) * 32,
                                    lds_addr_of(&As[(kk + 1) & 1][0]), K, M);
                __builtin_amdgcn_s_wait_tensorcnt(1);   // tile kk landed
            } else {
                __builtin_amdgcn_s_wait_tensorcnt(0);
            }
        }
        __syncthreads();   // publish tile kk to all waves

        Frag bf;
        {   // B fragment: lane = column (n0+nl), K-major contiguous
            const unsigned short* wrow = Wt + (size_t)(n0 + nl) * K + kk * 32;
            #pragma unroll
            for (int i = 0; i < 8; ++i)
                bf.u[i] = *(const unsigned int*)(wrow + pair_base(i, g));
        }
        const unsigned short* Abuf = &As[kk & 1][0];
        #pragma unroll
        for (int mt = 0; mt < 4; ++mt) {
            Frag af;
            const unsigned short* arow = Abuf + (mt * 16 + nl) * 32;
            #pragma unroll
            for (int i = 0; i < 8; ++i)
                af.u[i] = *(const unsigned int*)(arow + pair_base(i, g));
            acc[mt] = __builtin_amdgcn_wmma_f32_16x16x32_bf16(
                false, af.v, false, bf.v, (short)0, acc[mt], false, false);
        }
        __syncthreads();   // all waves done with buffer before it is re-DMAed
    }

    int col = n0 + nl;
    float bv = bias ? bias[col] : 0.0f;
    #pragma unroll
    for (int mt = 0; mt < 4; ++mt) {
        #pragma unroll
        for (int r = 0; r < 8; ++r) {
            int row = m0 + mt * 16 + r + 8 * g;
            float v = acc[mt][r] + bv;
            if (resid) v += resid[(size_t)row * N + col];
            if (act == 1) v = gelu_exact(v);
            if (outF) outF[(size_t)row * N + col] = v;
            if (outB) outB[(size_t)row * N + col] = f2bf(v);
        }
    }
}

// ---------------- fused attention per (b, h, 16-query tile) ----------------
// scores (WMMA) -> LDS fp32 -> masked softmax -> bf16 probs in LDS -> P*V (WMMA)

__global__ __launch_bounds__(128) void attn_kernel(const unsigned short* __restrict__ qB,
                                                   const unsigned short* __restrict__ kB,
                                                   const unsigned short* __restrict__ vB,
                                                   const int* __restrict__ xids,
                                                   unsigned short* __restrict__ ctxB,
                                                   float scale) {
    __shared__ float smask[512];
    __shared__ float sc[16 * 512];
    __shared__ unsigned short pb[16 * 512];
    __shared__ float red[16 * 8];
    __shared__ float rowstat[16];

    const int S = 512, Dm = 768, HS = 64;
    int qt = blockIdx.x;   // 0..31
    int h  = blockIdx.y;   // 0..11
    int b  = blockIdx.z;   // 0..15
    int tid = threadIdx.x;
    int wave = tid >> 5, lane = tid & 31;
    int g = lane >> 4, nl = lane & 15;

    for (int j = tid; j < S; j += 128)
        smask[j] = (xids[b * S + j] > 0) ? 0.0f : -1e9f;
    __syncthreads();

    // phase 1: scores = Q(16x64) * K^T, tiles of 16 keys; wave strides key tiles
    size_t qbase = (size_t)(b * S + qt * 16) * Dm + h * HS;
    for (int kt = wave; kt < 32; kt += 4) {
        v8f c = {};
        #pragma unroll
        for (int ks = 0; ks < 2; ++ks) {
            Frag a, bb;
            const unsigned short* arow = qB + qbase + (size_t)nl * Dm + ks * 32;
            const unsigned short* brow = kB + (size_t)(b * S + kt * 16 + nl) * Dm + h * HS + ks * 32;
            #pragma unroll
            for (int i = 0; i < 8; ++i) {
                a.u[i]  = *(const unsigned int*)(arow + pair_base(i, g));
                bb.u[i] = *(const unsigned int*)(brow + pair_base(i, g));
            }
            c = __builtin_amdgcn_wmma_f32_16x16x32_bf16(false, a.v, false, bb.v, (short)0, c, false, false);
        }
        #pragma unroll
        for (int r = 0; r < 8; ++r) {
            int m = r + 8 * g;
            int key = kt * 16 + nl;
            sc[m * 512 + key] = c[r] * scale + smask[key];
        }
    }
    __syncthreads();

    // phase 2: softmax over 512 keys per row (8 threads / row)
    {
        int row = tid >> 3, sub = tid & 7;
        float mx = -3.4e38f;
        for (int j = sub; j < S; j += 8) mx = fmaxf(mx, sc[row * 512 + j]);
        red[row * 8 + sub] = mx;
        __syncthreads();
        if (sub == 0) {
            float m2 = red[row * 8];
            #pragma unroll
            for (int i = 1; i < 8; ++i) m2 = fmaxf(m2, red[row * 8 + i]);
            rowstat[row] = m2;
        }
        __syncthreads();
        float rm = rowstat[row];
        float s = 0.0f;
        for (int j = sub; j < S; j += 8) {
            float e = expf(sc[row * 512 + j] - rm);
            sc[row * 512 + j] = e;
            s += e;
        }
        __syncthreads();
        red[row * 8 + sub] = s;
        __syncthreads();
        if (sub == 0) {
            float t2 = 0.0f;
            #pragma unroll
            for (int i = 0; i < 8; ++i) t2 += red[row * 8 + i];
            rowstat[row] = 1.0f / t2;
        }
        __syncthreads();
        float inv = rowstat[row];
        for (int j = sub; j < S; j += 8)
            pb[row * 512 + j] = f2bf(sc[row * 512 + j] * inv);
    }
    __syncthreads();

    // phase 3: ctx = P(16x512) * V(512x64); wave owns hs-slice wave*16
    v8f c = {};
    for (int ks = 0; ks < 16; ++ks) {
        Frag a, bb;
        const unsigned short* arow = pb + nl * 512 + ks * 32;
        #pragma unroll
        for (int i = 0; i < 8; ++i)
            a.u[i] = *(const unsigned int*)(arow + pair_base(i, g));
        int hs = h * HS + wave * 16 + nl;
        #pragma unroll
        for (int i = 0; i < 8; ++i) {
            int kidx = ks * 32 + pair_base(i, g);
            unsigned int lo = vB[(size_t)(b * S + kidx) * Dm + hs];
            unsigned int hi = vB[(size_t)(b * S + kidx + 1) * Dm + hs];
            bb.u[i] = lo | (hi << 16);
        }
        c = __builtin_amdgcn_wmma_f32_16x16x32_bf16(false, a.v, false, bb.v, (short)0, c, false, false);
    }
    #pragma unroll
    for (int r = 0; r < 8; ++r) {
        int m = r + 8 * g;
        size_t row = (size_t)(b * S + qt * 16 + m);
        ctxB[row * Dm + h * HS + wave * 16 + nl] = f2bf(c[r]);
    }
}

// ---------------- host side ----------------

extern "C" void kernel_launch(void* const* d_in, const int* in_sizes, int n_in,
                              void* d_out, int out_size, void* d_ws, size_t ws_size,
                              hipStream_t stream) {
    (void)in_sizes; (void)n_in; (void)out_size; (void)ws_size;
    const int L = 12, D = 768, FF = 3072, S = 512, B = 16;
    const int M = B * S;                 // 8192 rows
    const float SCALE = 0.03608439182435161f;   // 1/sqrt(768) per reference

    const int*   xids   = (const int*)  d_in[0];
    const float* wemb   = (const float*)d_in[1];
    const float* pemb   = (const float*)d_in[2];
    const float* ln_e_g = (const float*)d_in[3];
    const float* ln_e_b = (const float*)d_in[4];
    const float* ln1_g  = (const float*)d_in[5];
    const float* ln1_b  = (const float*)d_in[6];
    const float* ln2_g  = (const float*)d_in[7];
    const float* ln2_b  = (const float*)d_in[8];
    const float* Wq     = (const float*)d_in[9];
    const float* bq     = (const float*)d_in[10];
    const float* Wk     = (const float*)d_in[11];
    const float* bk     = (const float*)d_in[12];
    const float* Wv     = (const float*)d_in[13];
    const float* bv     = (const float*)d_in[14];
    const float* Wo     = (const float*)d_in[15];
    const float* bo     = (const float*)d_in[16];
    const float* W1     = (const float*)d_in[17];
    const float* b1     = (const float*)d_in[18];
    const float* W2     = (const float*)d_in[19];
    const float* b2     = (const float*)d_in[20];
    const float* poolW  = (const float*)d_in[21];
    const float* poolB  = (const float*)d_in[22];

    // workspace partition (256B aligned)
    char* ws = (char*)d_ws;
    size_t off = 0;
    auto alloc = [&](size_t bytes) { void* p = ws + off; off += (bytes + 255) & ~(size_t)255; return p; };
    unsigned short* wq_t  = (unsigned short*)alloc((size_t)L * D * D * 2);
    unsigned short* wk_t  = (unsigned short*)alloc((size_t)L * D * D * 2);
    unsigned short* wv_t  = (unsigned short*)alloc((size_t)L * D * D * 2);
    unsigned short* wo_t  = (unsigned short*)alloc((size_t)L * D * D * 2);
    unsigned short* w1_t  = (unsigned short*)alloc((size_t)L * D * FF * 2);
    unsigned short* w2_t  = (unsigned short*)alloc((size_t)L * FF * D * 2);
    unsigned short* pl_t  = (unsigned short*)alloc((size_t)D * D * 2);
    float*          hF    = (float*)        alloc((size_t)M * D * 4);
    unsigned short* hB    = (unsigned short*)alloc((size_t)M * D * 2);
    unsigned short* qB    = (unsigned short*)alloc((size_t)M * D * 2);
    unsigned short* kBuf  = (unsigned short*)alloc((size_t)M * D * 2);
    unsigned short* vBuf  = (unsigned short*)alloc((size_t)M * D * 2);
    unsigned short* ctxB  = (unsigned short*)alloc((size_t)M * D * 2);
    unsigned short* fB    = (unsigned short*)alloc((size_t)M * FF * 2);

    // ---- weight repack (fp32 -> bf16, N x K) ----
    {
        size_t t1 = (size_t)L * D * D;
        int g1 = (int)((t1 + 255) / 256);
        repack_headed<<<g1, 256, 0, stream>>>(Wq, wq_t, L);
        repack_headed<<<g1, 256, 0, stream>>>(Wk, wk_t, L);
        repack_headed<<<g1, 256, 0, stream>>>(Wv, wv_t, L);
        repack_plain<<<g1, 256, 0, stream>>>(Wo, wo_t, L, D, D);
        size_t t2 = (size_t)L * D * FF;
        int g2 = (int)((t2 + 255) / 256);
        repack_plain<<<g2, 256, 0, stream>>>(W1, w1_t, L, D, FF);
        repack_plain<<<g2, 256, 0, stream>>>(W2, w2_t, L, FF, D);
        size_t t3 = (size_t)D * D;
        int g3 = (int)((t3 + 255) / 256);
        repack_plain<<<g3, 256, 0, stream>>>(poolW, pl_t, 1, D, D);
    }

    // ---- embedding + LN(eps=1e-12) ----
    embed_ln_kernel<<<M, 256, 0, stream>>>(xids, wemb, pemb, ln_e_g, ln_e_b, hF, hB);

    dim3 gProj(M / 64, D / 64);
    dim3 gFF1(M / 64, FF / 64);
    dim3 aGrid(S / 16, 12, B);

    for (int l = 0; l < L; ++l) {
        // LN1 (in-place fp32, bf16 copy)
        ln_kernel<<<M, 256, 0, stream>>>(hF, hF, hB, ln1_g + l * D, ln1_b + l * D, 1e-5f);

        // QKV projections (bf16 outputs)
        gemm_bf16<<<gProj, 128, 0, stream>>>(hB, wq_t + (size_t)l * D * D, bq + l * D,
                                             nullptr, nullptr, qB, M, D, D, 0);
        gemm_bf16<<<gProj, 128, 0, stream>>>(hB, wk_t + (size_t)l * D * D, bk + l * D,
                                             nullptr, nullptr, kBuf, M, D, D, 0);
        gemm_bf16<<<gProj, 128, 0, stream>>>(hB, wv_t + (size_t)l * D * D, bv + l * D,
                                             nullptr, nullptr, vBuf, M, D, D, 0);

        // fused attention -> ctx bf16 in [b,s,h,e] layout
        attn_kernel<<<aGrid, 128, 0, stream>>>(qB, kBuf, vBuf, xids, ctxB, SCALE);

        // output projection + residual (residual = post-LN1 h, per reference)
        gemm_bf16<<<gProj, 128, 0, stream>>>(ctxB, wo_t + (size_t)l * D * D, bo + l * D,
                                             hF, hF, nullptr, M, D, D, 0);

        // LN2
        ln_kernel<<<M, 256, 0, stream>>>(hF, hF, hB, ln2_g + l * D, ln2_b + l * D, 1e-5f);

        // FFN
        gemm_bf16<<<gFF1, 128, 0, stream>>>(hB, w1_t + (size_t)l * D * FF, b1 + l * FF,
                                            nullptr, nullptr, fB, M, FF, D, 1 /*gelu*/);
        gemm_bf16<<<gProj, 128, 0, stream>>>(fB, w2_t + (size_t)l * FF * D, b2 + l * D,
                                             hF, hF, (l == L - 1) ? hB : nullptr, M, D, FF, 0);
    }

    // pooler: gelu(h * pool_W + pool_b) -> d_out (fp32)
    gemm_bf16<<<gProj, 128, 0, stream>>>(hB, pl_t, poolB, nullptr, (float*)d_out, nullptr,
                                         M, D, D, 1 /*gelu*/);
}
